// HypergraphTransformer_63771674411309
// MI455X (gfx1250) — matrix operand order repaired and checked by reading, not compiled, not run on hardware
//
#include <hip/hip_runtime.h>

// ---------------------------------------------------------------------------
// HypergraphTransformer forward for gfx1250 (MI455X).
// bf16 WMMA GEMM core with async Global->LDS tile staging (ASYNCcnt) and
// one-time weight packing into WMMA-fragment-major bf16 tiles.
// ---------------------------------------------------------------------------

typedef __attribute__((ext_vector_type(16))) __bf16  v16bf;
typedef __attribute__((ext_vector_type(8)))  float   v8f;
typedef __attribute__((ext_vector_type(4)))  float   v4f;
typedef __attribute__((ext_vector_type(4)))  unsigned int v4u;
typedef __attribute__((ext_vector_type(2)))  unsigned int v2u;

#define HGT_N   50000
#define HGT_E   1000
#define HGT_D   256
#define HGT_HD  4
#define HGT_M   30
#define HGT_L   3

#define BM 128
#define BN 128
#define BK 32
#define NJT 8                 // BN/16 WMMA col-tiles per wave
#define A_STRIDE 40           // ushorts per A row in LDS (32 + 8 pad)
#define BTILE 4096            // ushorts per packed B tile (32 x 128)

#define FLAG_ACC 1
#define FLAG_EXP 2

__device__ __host__ __forceinline__ unsigned short f32_to_bf16_bits(float f) {
  union { float f; unsigned int u; } c; c.f = f;
  unsigned int u = c.u;
  u += 0x7FFFu + ((u >> 16) & 1u);     // round-to-nearest-even
  return (unsigned short)(u >> 16);
}

union Frag16 { v16bf v; v4u q[2]; };
union CAcc   { v8f  v; float f[8]; };

__device__ __forceinline__ void async_copy_b128(unsigned lds_byte_addr,
                                                unsigned long long gaddr) {
  // GLOBAL_LOAD_ASYNC_TO_LDS_B128: memory -> LDS, tracked by ASYNCcnt.
  asm volatile("global_load_async_to_lds_b128 %0, %1, off"
               :: "v"(lds_byte_addr), "v"(gaddr) : "memory");
}
__device__ __forceinline__ void wait_asynccnt0() {
  asm volatile("s_wait_asynccnt 0x0" ::: "memory");
}

// ---------------------------------------------------------------------------
// Pack B[K,N] (f32, row-major, leading dim ldb) into WMMA fragment-major bf16:
// P[kt][nb][jt][ln][q], 4096 ushorts per (kt,nb) tile; zero-padded K/N edges.
// Fragment read for (jt,lane) is then 32 contiguous bytes.
// ---------------------------------------------------------------------------
__global__ void pack_b_kernel(const float* __restrict__ W,
                              unsigned short* __restrict__ P,
                              int K, int N, int ldb, int nkt, int nnb)
{
  size_t idx = (size_t)blockIdx.x * 256 + threadIdx.x;
  size_t total = (size_t)nkt * nnb * BTILE;
  if (idx >= total) return;
  int s  = (int)(idx & (BTILE - 1));
  int tb = (int)(idx >> 12);
  int nb = tb % nnb, kt = tb / nnb;
  int jt = s >> 9;
  int r  = s & 511;
  int ln = r >> 4, q = r & 15;
  int kr = q | ((ln >> 4) << 4);          // K row within 32-tile
  int col = nb * BN + jt * 16 + (ln & 15);
  int k   = kt * BK + kr;
  float v = (k < K && col < N) ? W[(size_t)k * ldb + col] : 0.0f;
  P[idx] = f32_to_bf16_bits(v);
}

// Packed projT: B[k][m] = proj[m][k] for m<30, zero otherwise. K=256, N=32->nnb=1.
__global__ void pack_proj_kernel(const float* __restrict__ proj,
                                 unsigned short* __restrict__ P)
{
  int idx = blockIdx.x * 256 + threadIdx.x;      // 8 * 4096
  if (idx >= 8 * BTILE) return;
  int s  = idx & (BTILE - 1);
  int kt = idx >> 12;
  int jt = s >> 9;
  int r  = s & 511;
  int ln = r >> 4, q = r & 15;
  int kr = q | ((ln >> 4) << 4);
  int col = jt * 16 + (ln & 15);
  int k   = kt * BK + kr;
  float v = (col < HGT_M) ? proj[(size_t)col * HGT_D + k] : 0.0f;
  P[idx] = f32_to_bf16_bits(v);
}

// ---------------------------------------------------------------------------
// GEMM: C[M,N] = op( A[M,K] @ Bpacked ), bf16 WMMA, f32 accumulate.
//   ABF16=true : A is bf16 row-major (K%32==0); tiles staged via async-to-LDS.
//   ABF16=false: A is f32 row-major (K%4==0);  register-staged + RNE convert.
// B is always pre-packed bf16 (pack_b_kernel layout); staged via async-to-LDS.
// flags: FLAG_ACC -> C += res + bias ; FLAG_EXP -> C = exp(alpha*res)+1e-6 (col<30 else 0)
// ---------------------------------------------------------------------------
template <bool ABF16>
__global__ __launch_bounds__(256) void wmma_gemm(
    const void* __restrict__ Ap, const unsigned short* __restrict__ Bp,
    const float* __restrict__ bias, float* __restrict__ C,
    int M, int N, int K, int lda, int ldc, int nnb,
    int flags, float alpha)
{
  __shared__ __align__(16) unsigned short Ash[2][BM * A_STRIDE];  // 2 x 10240 B
  __shared__ __align__(16) unsigned short Bsh[2][BTILE];          // 2 x  8192 B

  const int tid      = threadIdx.x;
  const int lane     = tid & 31;
  const int wave     = tid >> 5;
  const int rowBlock = blockIdx.x * BM;
  const int colBlock = blockIdx.y * BN;
  const int laneM    = lane & 15;
  const int laneHi   = lane >> 4;
  const int kTiles   = (K + BK - 1) / BK;

  CAcc acc[NJT];
  for (int j = 0; j < NJT; ++j)
    for (int r = 0; r < 8; ++r) acc[j].f[r] = 0.0f;

  // f32-A register staging coordinates (4 float4 per thread: 128 x 8 float4)
  int rA[4], cA[4];
  for (int i = 0; i < 4; ++i) {
    int f = i * 256 + tid;
    rA[i] = f >> 3;  cA[i] = (f & 7) * 4;
  }
  v4f aS[4];

  const float* Af32 = (const float*)Ap;
  const unsigned short* Abf = (const unsigned short*)Ap;

  // ---- tile staging helpers ----
  auto issueB = [&](int kt, int buf) {
    const unsigned short* src = Bp + ((size_t)(kt * nnb + blockIdx.y) * BTILE);
    for (int i = 0; i < 2; ++i) {
      int off = (i * 256 + tid) * 8;                       // 16-byte chunks
      async_copy_b128((unsigned)(size_t)&Bsh[buf][off],
                      (unsigned long long)(size_t)(src + off));
    }
  };
  auto issueA_bf16 = [&](int kt, int buf) {                // ABF16 path
    const int k0 = kt * BK;
    for (int i = 0; i < 2; ++i) {
      int f  = i * 256 + tid;                              // 512 chunks (128 x 4)
      int r  = f >> 2, c8 = (f & 3) * 8;
      int gr = rowBlock + r;
      if (gr < M) {
        async_copy_b128((unsigned)(size_t)&Ash[buf][r * A_STRIDE + c8],
                        (unsigned long long)(size_t)(Abf + (size_t)gr * lda + k0 + c8));
      }
    }
  };
  auto loadA_f32 = [&](int kt) {                           // f32 path, to regs
    const int k0 = kt * BK;
    for (int i = 0; i < 4; ++i) {
      aS[i] = (v4f){0.f, 0.f, 0.f, 0.f};
      int gr = rowBlock + rA[i], gk = k0 + cA[i];
      if (gr < M && gk < K) {
        const float* ap = Af32 + (size_t)gr * lda + gk;
        aS[i] = *(const v4f*)ap;
        if (gk + BK < K) __builtin_prefetch(ap + BK, 0, 1);
      }
    }
  };
  auto storeA_f32 = [&](int buf) {                         // regs -> LDS (bf16)
    for (int i = 0; i < 4; ++i) {
      unsigned int lo = (unsigned)f32_to_bf16_bits(aS[i].x) |
                        ((unsigned)f32_to_bf16_bits(aS[i].y) << 16);
      unsigned int hi = (unsigned)f32_to_bf16_bits(aS[i].z) |
                        ((unsigned)f32_to_bf16_bits(aS[i].w) << 16);
      v2u pk = {lo, hi};
      *(v2u*)&Ash[buf][rA[i] * A_STRIDE + cA[i]] = pk;     // ds_store_b64
    }
  };

  // ---- prologue: stage tile 0 into buffer 0 ----
  issueB(0, 0);
  if constexpr (ABF16) issueA_bf16(0, 0);
  else                 loadA_f32(0);

  int buf = 0;
  for (int kt = 0; kt < kTiles; ++kt) {
    if constexpr (!ABF16) storeA_f32(buf);
    wait_asynccnt0();            // tile kt resident in LDS
    __syncthreads();             // visible to all waves; prior reads of buf done

    if (kt + 1 < kTiles) {       // overlap next tile with this tile's WMMAs
      issueB(kt + 1, buf ^ 1);
      if constexpr (ABF16) issueA_bf16(kt + 1, buf ^ 1);
      else                 loadA_f32(kt + 1);
    }

    // A fragment: contiguous k = 16*laneHi + s (matches packed B); two b128 runs.
    Frag16 a;
    const unsigned short* arow = &Ash[buf][(wave * 16 + laneM) * A_STRIDE + laneHi * 16];
    a.q[0] = *(const v4u*)(arow);
    a.q[1] = *(const v4u*)(arow + 8);

    for (int j = 0; j < NJT; ++j) {
      Frag16 b;
      const unsigned short* bp = &Bsh[buf][j * 512 + lane * 16];
      b.q[0] = *(const v4u*)(bp);
      b.q[1] = *(const v4u*)(bp + 8);
      acc[j].v = __builtin_amdgcn_wmma_f32_16x16x32_bf16(
          false, a.v, false, b.v, (short)0, acc[j].v, false, false);
    }
    buf ^= 1;
  }

  // ---- epilogue ----
  const int waveRow0 = rowBlock + wave * 16;
  for (int j = 0; j < NJT; ++j) {
    int col = colBlock + j * 16 + laneM;
    if (col >= N) continue;
    float bcol = (bias != nullptr) ? bias[col] : 0.0f;
    for (int r = 0; r < 8; ++r) {
      int row = waveRow0 + laneHi * 8 + r;
      if (row >= M) continue;
      float vres = acc[j].f[r];
      float* cp = C + (size_t)row * ldc + col;
      if (flags & FLAG_EXP) {
        *cp = (col < HGT_M) ? (__expf(alpha * vres) + 1e-6f) : 0.0f;
      } else {
        vres += bcol;
        if (flags & FLAG_ACC) vres += *cp;
        *cp = vres;
      }
    }
  }
}

// z(bf16) = elu(layernorm(h) * g + b); one wave32 per row of 256.
__global__ __launch_bounds__(256) void ln_elu_kernel(
    const float* __restrict__ h, const float* __restrict__ g,
    const float* __restrict__ b, unsigned short* __restrict__ zbf, int nrows)
{
  int row  = blockIdx.x * 8 + (threadIdx.x >> 5);
  int lane = threadIdx.x & 31;
  if (row >= nrows) return;
  const float* hr = h + (size_t)row * HGT_D;
  float x[8]; float s = 0.f, ss = 0.f;
  for (int i = 0; i < 8; ++i) {
    x[i] = hr[lane + 32 * i];
    s += x[i]; ss += x[i] * x[i];
  }
  for (int o = 16; o > 0; o >>= 1) {
    s  += __shfl_xor(s,  o, 32);
    ss += __shfl_xor(ss, o, 32);
  }
  float mu  = s * (1.0f / HGT_D);
  float var = ss * (1.0f / HGT_D) - mu * mu;
  float rs  = rsqrtf(var + 1e-5f);
  unsigned short* zr = zbf + (size_t)row * HGT_D;
  for (int i = 0; i < 8; ++i) {
    int c = lane + 32 * i;
    float y = (x[i] - mu) * rs * g[c] + b[c];
    y = (y > 0.0f) ? y : (__expf(y) - 1.0f);
    zr[c] = f32_to_bf16_bits(y);
  }
}

__global__ void zero_kernel(float* __restrict__ p, int n)
{
  int i = blockIdx.x * 256 + threadIdx.x;
  if (i < n) p[i] = 0.0f;
}

// Split-K reduction: KV[h,m,d] += sum_n Kp[n,h,m] * V[n,h,d]; Ksum[h,m] += Kp.
#define KV_CHUNK 1024
__global__ __launch_bounds__(256) void kv_accum_kernel(
    const float* __restrict__ Kp, const float* __restrict__ V,
    float* __restrict__ KV, float* __restrict__ Ksum, int nrows)
{
  __shared__ float kp_sh[8 * 32];
  const int head = blockIdx.y;
  const int n0   = blockIdx.x * KV_CHUNK;
  const int nEnd = (n0 + KV_CHUNK < nrows) ? (n0 + KV_CHUNK) : nrows;
  const int t    = threadIdx.x;          // d = t (0..255)
  float acc[32];
  for (int m = 0; m < 32; ++m) acc[m] = 0.0f;

  for (int nb = n0; nb < nEnd; nb += 8) {
    int nn = nb + (t >> 5);
    int mm = t & 31;
    kp_sh[t] = (nn < nEnd) ? Kp[((size_t)nn * HGT_HD + head) * 32 + mm] : 0.0f;
    __syncthreads();
    int lim = nEnd - nb; if (lim > 8) lim = 8;
    for (int q = 0; q < lim; ++q) {
      float v = V[(((size_t)(nb + q)) * HGT_HD + head) * HGT_D + t];
      const float* kr = &kp_sh[q * 32];
      for (int m = 0; m < 32; ++m) acc[m] = fmaf(kr[m], v, acc[m]);
    }
    __syncthreads();
  }
  float* kvh = KV + (size_t)head * 32 * HGT_D;
  for (int m = 0; m < 32; ++m) atomicAdd(&kvh[m * HGT_D + t], acc[m]);
  if (t < 32) {
    float s = 0.0f;
    for (int n = n0; n < nEnd; ++n) s += Kp[((size_t)n * HGT_HD + head) * 32 + t];
    atomicAdd(&Ksum[head * 32 + t], s);
  }
}

// Z[n,h,d] = (Qp[n,h,:]·KV[h,:,d]) / (Qp[n,h,:]·Ksum[h,:] + 1e-6)
__global__ __launch_bounds__(256) void attn_out_kernel(
    const float* __restrict__ Qp, const float* __restrict__ KV,
    const float* __restrict__ Ksum, float* __restrict__ Z, int nrows)
{
  size_t e = (size_t)blockIdx.x * 256 + threadIdx.x;   // over N*1024
  if (e >= (size_t)nrows * (HGT_HD * HGT_D)) return;
  int d      = (int)(e & (HGT_D - 1));
  size_t rh  = e >> 8;                                  // n*4 + h
  int head   = (int)(rh & (HGT_HD - 1));
  const float* qp = Qp + rh * 32;
  const float* kv = KV + (size_t)head * 32 * HGT_D + d;
  const float* ks = Ksum + head * 32;
  float num = 0.0f, den = 0.0f;
  for (int m = 0; m < 32; ++m) {
    float q = qp[m];
    num = fmaf(q, kv[(size_t)m * HGT_D], num);
    den = fmaf(q, ks[m], den);
  }
  Z[e] = num / (den + 1e-6f);
}

static inline int cdiv_i(int a, int b) { return (a + b - 1) / b; }

extern "C" void kernel_launch(void* const* d_in, const int* in_sizes, int n_in,
                              void* d_out, int out_size, void* d_ws, size_t ws_size,
                              hipStream_t stream) {
  const float* x    = (const float*)d_in[0];
  const float* Hm   = (const float*)d_in[1];
  const float* proj = (const float*)d_in[2];
  const float* W_in = (const float*)d_in[3];
  const float* b_in = (const float*)d_in[4];
  const float* W_he = (const float*)d_in[5];
  const float* b_he = (const float*)d_in[6];
  const float* ln_g = (const float*)d_in[7];
  const float* ln_b = (const float*)d_in[8];
  const float* Wq   = (const float*)d_in[9];
  const float* bq   = (const float*)d_in[10];
  const float* Wk   = (const float*)d_in[11];
  const float* bk   = (const float*)d_in[12];
  const float* Wv   = (const float*)d_in[13];
  const float* bv   = (const float*)d_in[14];
  const float* Wo   = (const float*)d_in[15];
  const float* bo   = (const float*)d_in[16];
  (void)ws_size; (void)n_in; (void)in_sizes;

  const int N = HGT_N, D = HGT_D, DH = HGT_HD * HGT_D;   // DH = 1024

  // ---- workspace layout (byte-based) ----
  char* wsb = (char*)d_ws;
  size_t off = 0;
  auto alloc_f = [&](size_t n) { float* p = (float*)(wsb + off);
                                 off += n * sizeof(float); return p; };
  auto alloc_u = [&](size_t n) { unsigned short* p = (unsigned short*)(wsb + off);
                                 off += ((n * 2 + 15) & ~(size_t)15); return p; };

  float* hbuf = alloc_f((size_t)N * D);
  float* Qbuf = alloc_f((size_t)N * DH);    // reused as Z after Qp is built
  float* Kbuf = alloc_f((size_t)N * DH);
  float* Vbuf = alloc_f((size_t)N * DH);
  float* Qp   = alloc_f((size_t)N * HGT_HD * 32);
  float* Kp   = alloc_f((size_t)N * HGT_HD * 32);
  float* KVb  = alloc_f((size_t)HGT_HD * 32 * D);  // KV then Ksum, contiguous
  float* Ksum = alloc_f((size_t)HGT_HD * 32);
  unsigned short* zbf   = alloc_u((size_t)N * D);
  // packed bf16 B operands
  const int nktD  = D  / BK;            // 8   (K=256)
  const int nktE  = cdiv_i(HGT_E, BK);  // 32  (K=1000)
  const int nktDH = DH / BK;            // 32  (K=1024)
  unsigned short* W_inP = alloc_u((size_t)nktD  * 2 * BTILE);
  unsigned short* W_heP = alloc_u((size_t)nktE  * 2 * BTILE);
  unsigned short* projP = alloc_u((size_t)nktD  * 1 * BTILE);
  unsigned short* WqP[HGT_L]; unsigned short* WkP[HGT_L];
  unsigned short* WvP[HGT_L]; unsigned short* WoP[HGT_L];
  for (int l = 0; l < HGT_L; ++l) {
    WqP[l] = alloc_u((size_t)nktD  * 8 * BTILE);
    WkP[l] = alloc_u((size_t)nktD  * 8 * BTILE);
    WvP[l] = alloc_u((size_t)nktD  * 8 * BTILE);
    WoP[l] = alloc_u((size_t)nktDH * 2 * BTILE);
  }

  const float alpha = 0.25f / sqrtf((float)HGT_M);   // d^-1/4 * m^-1/2
  dim3 blk(256);

  // ---- one-time weight packing (f32 -> fragment-major bf16) ----
  pack_b_kernel<<<cdiv_i(nktD * 2 * BTILE, 256), blk, 0, stream>>>(
      W_in, W_inP, D, D, D, nktD, 2);
  pack_b_kernel<<<cdiv_i(nktE * 2 * BTILE, 256), blk, 0, stream>>>(
      W_he, W_heP, HGT_E, D, D, nktE, 2);
  pack_proj_kernel<<<cdiv_i(nktD * BTILE, 256), blk, 0, stream>>>(proj, projP);
  for (int l = 0; l < HGT_L; ++l) {
    pack_b_kernel<<<cdiv_i(nktD * 8 * BTILE, 256), blk, 0, stream>>>(
        Wq + (size_t)l * D * DH, WqP[l], D, DH, DH, nktD, 8);
    pack_b_kernel<<<cdiv_i(nktD * 8 * BTILE, 256), blk, 0, stream>>>(
        Wk + (size_t)l * D * DH, WkP[l], D, DH, DH, nktD, 8);
    pack_b_kernel<<<cdiv_i(nktD * 8 * BTILE, 256), blk, 0, stream>>>(
        Wv + (size_t)l * D * DH, WvP[l], D, DH, DH, nktD, 8);
    pack_b_kernel<<<cdiv_i(nktDH * 2 * BTILE, 256), blk, 0, stream>>>(
        Wo + (size_t)l * DH * D, WoP[l], DH, D, D, nktDH, 2);
  }

  dim3 gEmb(cdiv_i(N, BM), 2);                  // 391 x 2  (N-out = 256)
  dim3 gQKV(cdiv_i(N, BM), 8);                  // 391 x 8  (N-out = 1024)
  dim3 gFeat(cdiv_i(N * HGT_HD, BM), 1);        // 1563 x 1 (N-out = 32)

  // h = x @ W_in + b_in ; h += H @ W_he + b_he
  wmma_gemm<false><<<gEmb, blk, 0, stream>>>(x, W_inP, b_in, hbuf,
      N, D, D, D, D, 2, 0, 0.0f);
  wmma_gemm<false><<<gEmb, blk, 0, stream>>>(Hm, W_heP, b_he, hbuf,
      N, D, HGT_E, HGT_E, D, 2, FLAG_ACC, 0.0f);

  for (int l = 0; l < HGT_L; ++l) {
    // z(bf16) = elu(layernorm(h))
    ln_elu_kernel<<<cdiv_i(N, 8), blk, 0, stream>>>(hbuf, ln_g + l * D,
                                                    ln_b + l * D, zbf, N);
    // Q/K/V = z @ W + b   (bf16 A, fully async-staged)
    wmma_gemm<true><<<gQKV, blk, 0, stream>>>(zbf, WqP[l], bq + (size_t)l * DH,
        Qbuf, N, DH, D, D, DH, 8, 0, 0.0f);
    wmma_gemm<true><<<gQKV, blk, 0, stream>>>(zbf, WkP[l], bk + (size_t)l * DH,
        Kbuf, N, DH, D, D, DH, 8, 0, 0.0f);
    wmma_gemm<true><<<gQKV, blk, 0, stream>>>(zbf, WvP[l], bv + (size_t)l * DH,
        Vbuf, N, DH, D, D, DH, 8, 0, 0.0f);

    // FAVOR+ feature maps: Qp/Kp = exp(alpha * Q @ projT) + 1e-6 (m padded to 32)
    wmma_gemm<false><<<gFeat, blk, 0, stream>>>(Qbuf, projP, nullptr, Qp,
        N * HGT_HD, 32, D, D, 32, 1, FLAG_EXP, alpha);
    wmma_gemm<false><<<gFeat, blk, 0, stream>>>(Kbuf, projP, nullptr, Kp,
        N * HGT_HD, 32, D, D, 32, 1, FLAG_EXP, alpha);

    // KV[h,m,d], Ksum[h,m] via split-K atomics
    int kvN = HGT_HD * 32 * D + HGT_HD * 32;
    zero_kernel<<<cdiv_i(kvN, 256), blk, 0, stream>>>(KVb, kvN);
    dim3 gKV(cdiv_i(N, KV_CHUNK), HGT_HD);
    kv_accum_kernel<<<gKV, blk, 0, stream>>>(Kp, Vbuf, KVb, Ksum, N);

    // Z = (Qp@KV) / (Qp@Ksum + 1e-6); write into Qbuf (Q is dead)
    attn_out_kernel<<<cdiv_i(N * DH, 256), blk, 0, stream>>>(Qp, KVb, Ksum, Qbuf, N);

    // h = h + Z @ Wo + bo   (residual via accumulate)
    wmma_gemm<false><<<gEmb, blk, 0, stream>>>(Qbuf, WoP[l], bo + (size_t)l * D,
        hbuf, N, D, DH, DH, D, 2, FLAG_ACC, 0.0f);
  }

  hipMemcpyAsync(d_out, hbuf, (size_t)out_size * sizeof(float),
                 hipMemcpyDeviceToDevice, stream);
}
